// VQ_7576322310629
// MI455X (gfx1250) — matrix operand order repaired and checked by reading, not compiled, not run on hardware
//
#include <hip/hip_runtime.h>

// ---------------------------------------------------------------------------
// VQ-VAE quantization for MI455X (gfx1250, wave32, WMMA).
//   sim GEMM: f16 hi/lo split, V_WMMA_F32_16X16X32_F16 (3 terms, f32 accum)
//   codebook staged in LDS pre-swizzled to the B-matrix wave layout
//   outputs (gather + loss) computed in exact f32 -> precision only affects
//   argmin, where error (~2^-23 rel) << typical code gap (~1e-3).
// ---------------------------------------------------------------------------

typedef __attribute__((ext_vector_type(16))) _Float16 v16h;
typedef __attribute__((ext_vector_type(8)))  float    v8f;
typedef __attribute__((ext_vector_type(4)))  float    v4f;

#define VQ_D 64
#define VQ_K 512
#define VQ_ROWS (32 * 64 * 64)          // 131072
#define VQ_OUT_ELEMS (VQ_ROWS * VQ_D)   // 8388608

// LDS layout (bytes), dynamic shared memory:
//   embT : float [512][68]  (transposed codebook, padded; col 64 = ||e||^2)
//   ehi  : _Float16, B-layout slots [(khalf*512 + n)*16 + e]
//   elo  : _Float16, same layout (f16 residuals)
#define LDS_EMBT_BYTES (VQ_K * 68 * 4)      // 139264
#define LDS_HILO_BYTES (4 * VQ_K * 16 * 2)  // 65536 each
#define LDS_HI_OFF  LDS_EMBT_BYTES
#define LDS_LO_OFF  (LDS_HI_OFF + LDS_HILO_BYTES)
#define LDS_TOTAL   (LDS_LO_OFF + LDS_HILO_BYTES)  // 270336 B < 320 KB/WGP

__global__ void vq_zero_loss_kernel(float* loss) {
    if (threadIdx.x == 0) *loss = 0.0f;
}

__global__ __launch_bounds__(256)
void vq_main_kernel(const float* __restrict__ x,
                    const float* __restrict__ emb,
                    float* __restrict__ out,
                    float* __restrict__ loss)
{
    extern __shared__ char smem[];
    float*    embT = reinterpret_cast<float*>(smem);
    _Float16* ehi  = reinterpret_cast<_Float16*>(smem + LDS_HI_OFF);
    _Float16* elo  = reinterpret_cast<_Float16*>(smem + LDS_LO_OFF);

    const int tid = threadIdx.x;

    // ---- Stage codebook into LDS (coalesced float4 reads; L2-resident) ----
    const v4f* emb4 = reinterpret_cast<const v4f*>(emb);
    for (int i4 = tid; i4 < (VQ_D * VQ_K) / 4; i4 += 256) {
        const int i = i4 * 4;
        const int d = i >> 9;          // emb row (0..63)
        const int n = i & 511;         // emb col base (multiple of 4)
        const int khalf = (d >> 5) * 2 + ((d >> 3) & 1);  // (k-step, lane half) slot block (0..3)
        const int e     = (d & 7) | ((d >> 1) & 8);       // table-order element within lane half
        v4f v = emb4[i4];
#pragma unroll
        for (int j = 0; j < 4; ++j) {
            const float vv = v[j];
            embT[(n + j) * 68 + d] = vv;                 // transposed f32 copy
            const _Float16 h = (_Float16)vv;
            const int slot = (khalf * 512 + (n + j)) * 16 + e;
            ehi[slot] = h;
            elo[slot] = (_Float16)(vv - (float)h);       // f16 residual
        }
    }
    __syncthreads();

    // ---- ||e_k||^2 into the padding column ----
    for (int n = tid; n < VQ_K; n += 256) {
        float s = 0.0f;
#pragma unroll 8
        for (int d2 = 0; d2 < VQ_D; ++d2) {
            const float v = embT[n * 68 + d2];
            s += v * v;
        }
        embT[n * 68 + 64] = s;
    }
    __syncthreads();

    // ---- Per-wave setup: 16 rows per wave, 128 rows per block ----
    const int wave  = tid >> 5;
    const int lane  = tid & 31;
    const int lhalf = lane >> 4;       // 0 / 1
    const int l16   = lane & 15;
    const int myrow = blockIdx.x * 128 + wave * 16 + l16;
    const int off   = lhalf * 8;       // A-layout K sub-offset for this half

    // Load A tile: x[myrow][*]. 16-bit A layout (16x32 per k-step):
    //   e=0..7  -> K = off + e ;  e=8..15 -> K = 16 + off + (e-8)
    const float* xrow = x + (size_t)myrow * VQ_D;
    v4f xr[8];  // exact f32 copy kept for out/loss (x read from HBM once)
#pragma unroll
    for (int ks = 0; ks < 2; ++ks) {
        xr[ks * 4 + 0] = *reinterpret_cast<const v4f*>(xrow + ks * 32 + off + 0);
        xr[ks * 4 + 1] = *reinterpret_cast<const v4f*>(xrow + ks * 32 + off + 4);
        xr[ks * 4 + 2] = *reinterpret_cast<const v4f*>(xrow + ks * 32 + off + 16);
        xr[ks * 4 + 3] = *reinterpret_cast<const v4f*>(xrow + ks * 32 + off + 20);
    }
    v16h ah[2], al[2];
#pragma unroll
    for (int ks = 0; ks < 2; ++ks) {
#pragma unroll
        for (int e = 0; e < 16; ++e) {
            const float v = xr[ks * 4 + (e >> 2)][e & 3];
            const _Float16 h = (_Float16)v;
            ah[ks][e] = h;
            al[ks][e] = (_Float16)(v - (float)h);
        }
    }

    // ---- Distance scan over 32 code tiles of 16 ----
    float minv[8];
    int   mini[8];
#pragma unroll
    for (int v = 0; v < 8; ++v) { minv[v] = 3.0e38f; mini[v] = 0; }

#pragma unroll 2
    for (int t = 0; t < 32; ++t) {
        v8f acc = {};
#pragma unroll
        for (int ks = 0; ks < 2; ++ks) {
            const int khalf = ks * 2 + lhalf;                // lanes16-31 -> K+8
            const int slot  = (khalf * 512 + t * 16 + l16) * 16;
            const v16h bh = *reinterpret_cast<const v16h*>(ehi + slot);
            const v16h bl = *reinterpret_cast<const v16h*>(elo + slot);
            // sim ~= xh*eh + xh*el + xl*eh  (f32 accumulate)
            acc = __builtin_amdgcn_wmma_f32_16x16x32_f16(false, ah[ks], false, bh,
                                                         (short)0, acc, false, false);
            acc = __builtin_amdgcn_wmma_f32_16x16x32_f16(false, ah[ks], false, bl,
                                                         (short)0, acc, false, false);
            acc = __builtin_amdgcn_wmma_f32_16x16x32_f16(false, al[ks], false, bh,
                                                         (short)0, acc, false, false);
        }
        const int   code = t * 16 + l16;
        const float e2   = embT[code * 68 + 64];             // broadcast-friendly
#pragma unroll
        for (int v = 0; v < 8; ++v) {
            // dist = ||e||^2 - 2*sim   (||f||^2 is row-constant: irrelevant)
            const float dd = e2 - 2.0f * acc[v];
            if (dd < minv[v]) { minv[v] = dd; mini[v] = code; }  // first-min tiebreak
        }
    }

    // ---- Argmin reduction across the 16-lane half (lexicographic on (val,idx)) ----
#pragma unroll
    for (int v = 0; v < 8; ++v) {
        float bv = minv[v];
        int   bi = mini[v];
#pragma unroll
        for (int m = 1; m < 16; m <<= 1) {
            const float ov = __shfl_xor(bv, m, 32);
            const int   oi = __shfl_xor(bi, m, 32);
            if (ov < bv || (ov == bv && oi < bi)) { bv = ov; bi = oi; }
        }
        minv[v] = bv; mini[v] = bi;
    }

    // Row r lives in element (r&7) of half (r<8 ? lanes0-15 : lanes16-31).
    // Broadcast so every lane learns the index for its write-row (= l16).
    int myidx = 0;
#pragma unroll
    for (int r = 0; r < 16; ++r) {
        const int bi = __shfl(mini[r & 7], (r < 8) ? 0 : 16, 32);
        if (l16 == r) myidx = bi;
    }

    // ---- Gather quantized column, write out = x + (q - x), accumulate loss ----
    const float* qcol = embT + myidx * 68;   // contiguous per lane; banks ~ random
    float* orow = out + (size_t)myrow * VQ_D;
    float lsum = 0.0f;
#pragma unroll
    for (int p = 0; p < 8; ++p) {
        const int base = (p >> 2) * 32 + off + ((p >> 1) & 1) * 16 + (p & 1) * 4;
        const v4f xv = xr[p];
        v4f ov;
#pragma unroll
        for (int j = 0; j < 4; ++j) {
            const float q  = qcol[base + j];
            const float dt = q - xv[j];
            lsum += dt * dt;
            ov[j] = xv[j] + dt;              // straight-through, mimics reference fp order
        }
        *reinterpret_cast<v4f*>(orow + base) = ov;
    }

    // loss = (BETA + 1) * mean((q-x)^2) = 1.25 * sum / N
#pragma unroll
    for (int m = 1; m < 32; m <<= 1) lsum += __shfl_xor(lsum, m, 32);
    if (lane == 0) atomicAdd(loss, lsum * (1.25f / (float)VQ_OUT_ELEMS));
}

extern "C" void kernel_launch(void* const* d_in, const int* in_sizes, int n_in,
                              void* d_out, int out_size, void* d_ws, size_t ws_size,
                              hipStream_t stream) {
    (void)in_sizes; (void)n_in; (void)d_ws; (void)ws_size; (void)out_size;
    const float* x   = reinterpret_cast<const float*>(d_in[0]);
    const float* emb = reinterpret_cast<const float*>(d_in[1]);
    float* out  = reinterpret_cast<float*>(d_out);
    float* loss = out + VQ_OUT_ELEMS;

    // Opt into >default dynamic LDS (264 KB of the 320 KB WGP pool). Idempotent.
    hipFuncSetAttribute(reinterpret_cast<const void*>(vq_main_kernel),
                        hipFuncAttributeMaxDynamicSharedMemorySize, LDS_TOTAL);

    vq_zero_loss_kernel<<<1, 32, 0, stream>>>(loss);
    vq_main_kernel<<<dim3(VQ_ROWS / 128), dim3(256), LDS_TOTAL, stream>>>(x, emb, out, loss);
}
